// TransRppg_25323127177427
// MI455X (gfx1250) — matrix operand order repaired and checked by reading, not compiled, not run on hardware
//
#include <hip/hip_runtime.h>
#include <math.h>

// ---------------- types ----------------
typedef __attribute__((ext_vector_type(16))) __bf16 bf16x16;
typedef __attribute__((ext_vector_type(8)))  float  f32x8;
typedef unsigned int u32x4 __attribute__((ext_vector_type(4)));
typedef int          i32x4 __attribute__((ext_vector_type(4)));
typedef int          i32x8 __attribute__((ext_vector_type(8)));

struct __align__(16) U128 { unsigned int w[4]; };
union Frag { U128 q[2]; bf16x16 v; };

#define HAVE_TDM (__has_builtin(__builtin_amdgcn_tensor_load_to_lds) && \
                  __has_builtin(__builtin_amdgcn_s_wait_tensorcnt))

__device__ __forceinline__ unsigned short f2bf(float f) {
  unsigned int u = __float_as_uint(f);
  if ((u & 0x7F800000u) == 0x7F800000u) return (unsigned short)(u >> 16); // inf/nan
  u += 0x7FFFu + ((u >> 16) & 1u);                                        // RNE
  return (unsigned short)(u >> 16);
}

__device__ __forceinline__ float block_sum(float v, float* red) {
  int tid = threadIdx.x;
  red[tid] = v; __syncthreads();
  for (int o = 128; o > 0; o >>= 1) { if (tid < o) red[tid] += red[tid + o]; __syncthreads(); }
  float r = red[0]; __syncthreads();
  return r;
}
__device__ __forceinline__ float block_max(float v, float* red) {
  int tid = threadIdx.x;
  red[tid] = v; __syncthreads();
  for (int o = 128; o > 0; o >>= 1) { if (tid < o) red[tid] = fmaxf(red[tid], red[tid + o]); __syncthreads(); }
  float r = red[0]; __syncthreads();
  return r;
}

#if HAVE_TDM
// Issue one TDM load of a 32(k) x 128(n) bf16 tile of Wt[N][Kp] into LDS,
// padded to the 80B row stride the fragment loader consumes.
__device__ __forceinline__ void tdm_issue_b(const unsigned short* Wt, int Kp, int N,
                                            int bn0, int k0, unsigned ldsAddr) {
  unsigned long long ga = (unsigned long long)(const void*)Wt +
      ((unsigned long long)(unsigned)bn0 * (unsigned)Kp + (unsigned)k0) * 2ull;
  u32x4 g0;
  g0[0] = 1u;                                                   // count=1
  g0[1] = ldsAddr;                                              // lds_addr
  g0[2] = (unsigned)ga;                                         // global_addr[31:0]
  g0[3] = (unsigned)((ga >> 32) & 0x01FFFFFFull) | 0x80000000u; // [56:32] | type=2
  i32x8 g1;
  g1[0] = (int)((1u << 16)    // data_size = 2B
              | (1u << 20)    // pad_enable
              | (3u << 22)    // pad_interval: every 16 DWORDs (64B)
              | (3u << 25));  // pad_amount: 4 DWORDs (16B)
  g1[1] = (int)(((unsigned)Kp & 0xFFFFu) << 16);  // tensor_dim0
  g1[2] = (int)(((unsigned)N  & 0xFFFFu) << 16);  // tensor_dim1
  g1[3] = (int)(32u << 16);                       // tile_dim0 = 32 (k)
  g1[4] = 128;                                    // tile_dim1 = 128 (n rows)
  g1[5] = Kp;                                     // tensor_dim0_stride
  g1[6] = 0; g1[7] = 0;
  i32x4 z4 = (i32x4)0;
#if defined(__clang_major__) && __clang_major__ >= 23
  i32x8 z8 = (i32x8)0;
  __builtin_amdgcn_tensor_load_to_lds(g0, g1, z4, z4, z8, 0);
#else
  __builtin_amdgcn_tensor_load_to_lds(g0, g1, z4, z4, 0);
#endif
}
#endif

// ---------------- generic strided WMMA GEMM ----------------
// Block tile 64(m) x 128(n) x 32(k); 8 waves in 2x4; each wave a 32x32 slab:
// 2 A-frags x 2 B-frags -> 4 WMMAs per K step.  Weights (flags&8) arrive as
// pre-transposed bf16 Wt[N][Kp] via double-buffered TDM; strided f32 B
// (attention) is staged through VGPRs with f32->bf16 conversion.
struct GemmP {
  const float* A;  long long aOff, aSB, aSH; int sAr, sAc;
  const float* Bm; long long bOff, bSB, bSH; int sBr, sBc;
  const unsigned short* Wt; int Kp;        // bf16^T weights (flags&8)
  const float* bias;
  const float* R;  long long rOff, rSB, rSH; int ldr;
  float* C;        long long cOff, cSB, cSH; int ldc;
  int M, N, K, H, flags;  // flags: 1=bias 2=gelu 4=residual 8=Wt/TDM
};

__global__ __launch_bounds__(256) void wmma_gemm_k(GemmP p) {
  __shared__ __align__(16) unsigned short As[64 * 40];       // 64 m-rows
  __shared__ __align__(16) unsigned short Bs[2 * 128 * 40];  // 2 x 128 n-rows

  const int t = threadIdx.x;
  const int lane = t & 31;
  const int wave = t >> 5;
  const int mw = wave >> 2;   // 0..1 : 32-row slab
  const int nw = wave & 3;    // 0..3 : 32-col slab
  const int z = blockIdx.z;
  const int zb = z / p.H, zh = z % p.H;
  const float* Ab = p.A + p.aOff + (long long)zb * p.aSB + (long long)zh * p.aSH;
  const float* Bb = (p.flags & 8) ? nullptr
      : (p.Bm + p.bOff + (long long)zb * p.bSB + (long long)zh * p.bSH);
  const int bm0 = blockIdx.y << 6;
  const int bn0 = blockIdx.x << 7;
  const int ktiles = (p.K + 31) >> 5;
  const bool bFastK = (p.sBc > p.sBr);
  const bool useTDM = (p.flags & 8) != 0;

  f32x8 a00 = {}, a01 = {}, a10 = {}, a11 = {};

  const int am  = t >> 2;          // 0..63
  const int ak0 = (t & 3) << 3;    // 0,8,16,24
  const int gmA = bm0 + am;
  const int gmC = (gmA < p.M) ? gmA : (p.M - 1);
  const float* aRow = Ab + (long long)gmC * p.sAr;
  const bool aRowIn = (gmA < p.M);

  const unsigned ldsB0 = (unsigned)(unsigned long long)(const void*)&Bs[0];
#if HAVE_TDM
  if (useTDM && t < 32) tdm_issue_b(p.Wt, p.Kp, p.N, bn0, 0, ldsB0);  // prologue tile 0
#endif

  for (int kt = 0; kt < ktiles; ++kt) {
    const int k0 = kt << 5;
    const int cur = kt & 1;
    const int bufOff = cur * (128 * 40);
    __syncthreads();

    // ---- stage B tile ----
    if (p.flags & 8) {
#if HAVE_TDM
      if (t < 32 && kt + 1 < ktiles)  // async: fetch next tile into other buffer
        tdm_issue_b(p.Wt, p.Kp, p.N, bn0, k0 + 32,
                    ldsB0 + (unsigned)((cur ^ 1) * (128 * 40 * 2)));
#else
      { // manual fallback: Wt rows are 16B aligned (Kp multiple of 16)
        const int bn  = t >> 1;
        const int bk0 = (t & 1) << 4;
        const int gn  = bn0 + bn;
        const int gnc = (gn < p.N) ? gn : (p.N - 1);
        #pragma unroll
        for (int h = 0; h < 2; ++h) {
          const int kk = k0 + bk0 + h * 8;
          U128 u = *(const U128*)(p.Wt + (long long)gnc * p.Kp + (kk < p.Kp ? kk : 0));
          if (gn >= p.N || kk >= p.Kp) { u.w[0] = u.w[1] = u.w[2] = u.w[3] = 0; }
          *(U128*)&Bs[bufOff + bn * 40 + bk0 + h * 8] = u;
        }
      }
#endif
    } else if (bFastK) {  // f32 B contiguous-ish along k (Q@K^T): b128 LDS stores
      const int bn  = t >> 1;          // 0..127
      const int bk0 = (t & 1) << 4;    // 0 or 16
      const int gn  = bn0 + bn;
      const int gnc = (gn < p.N) ? gn : (p.N - 1);
      #pragma unroll
      for (int h = 0; h < 2; ++h) {
        __align__(16) unsigned short tmp[8];
        #pragma unroll
        for (int j = 0; j < 8; ++j) {
          const int gk  = k0 + bk0 + h * 8 + j;
          const int gkc = (gk < p.K) ? gk : (p.K - 1);
          float v = Bb[(long long)(gkc * p.sBr + gnc * p.sBc)];  // clamped
          v = (gn < p.N && gk < p.K) ? v : 0.0f;
          tmp[j] = f2bf(v);
        }
        *(U128*)&Bs[bufOff + bn * 40 + bk0 + h * 8] = *(const U128*)tmp;
      }
    } else {              // f32 B contiguous along n (att@V): scatter b16 stores
      const int bk  = t >> 3;          // 0..31
      const int n0t = (t & 7) << 4;    // 0..112
      const int gk  = k0 + bk;
      const int gkc = (gk < p.K) ? gk : (p.K - 1);
      const float* bRow = Bb + (long long)gkc * p.sBr;
      #pragma unroll
      for (int j = 0; j < 16; ++j) {
        const int gn  = bn0 + n0t + j;
        const int gnc = (gn < p.N) ? gn : (p.N - 1);
        float v = bRow[(long long)gnc * p.sBc];
        v = (gn < p.N && gk < p.K) ? v : 0.0f;
        Bs[bufOff + (n0t + j) * 40 + bk] = f2bf(v);
      }
    }

    // ---- stage A tile (64 x 32), f32 -> bf16 (overlaps the TDM transfer) ----
    {
      __align__(16) unsigned short tmp[8];
      const long long eBase = (long long)gmC * p.sAr + k0 + ak0;
      if (p.sAc == 1 && aRowIn && (k0 + 32) <= p.K && ((eBase & 3) == 0)) {
        const float4 f0 = *(const float4*)(aRow + k0 + ak0);
        const float4 f1 = *(const float4*)(aRow + k0 + ak0 + 4);
        tmp[0] = f2bf(f0.x); tmp[1] = f2bf(f0.y); tmp[2] = f2bf(f0.z); tmp[3] = f2bf(f0.w);
        tmp[4] = f2bf(f1.x); tmp[5] = f2bf(f1.y); tmp[6] = f2bf(f1.z); tmp[7] = f2bf(f1.w);
      } else {
        #pragma unroll
        for (int j = 0; j < 8; ++j) {
          const int gk  = k0 + ak0 + j;
          const int gkc = (gk < p.K) ? gk : (p.K - 1);
          float v = aRow[gkc * p.sAc];                 // clamped, branchless
          v = (aRowIn && gk < p.K) ? v : 0.0f;
          tmp[j] = f2bf(v);
        }
      }
      *(U128*)&As[am * 40 + ak0] = *(const U128*)tmp;
      if (aRowIn && (k0 + 32) < p.K)                   // global_prefetch next K tile
        __builtin_prefetch(&aRow[(k0 + 32 + ak0) * p.sAc], 0, 3);
    }

#if HAVE_TDM
    if (useTDM && t < 32) {  // TDM completes in order: <=1 left => tile kt landed
      if (kt + 1 < ktiles) __builtin_amdgcn_s_wait_tensorcnt(1);
      else                 __builtin_amdgcn_s_wait_tensorcnt(0);
    }
#endif
    __syncthreads();

    // ---- fragments per wave32 VGPR layouts, 4 WMMAs / wave ----
    const int lm  = lane & 15;
    const int akb = (lane & 16) ? 8 : 0;
    const int bkb = (lane & 16) ? 8 : 0;
    Frag fa0, fa1, fb0, fb1;
    const int ar0 = (mw * 32 + lm) * 40;
    const int ar1 = ar0 + 16 * 40;
    fa0.q[0] = *(const U128*)&As[ar0 + akb];
    fa0.q[1] = *(const U128*)&As[ar0 + 16 + akb];
    fa1.q[0] = *(const U128*)&As[ar1 + akb];
    fa1.q[1] = *(const U128*)&As[ar1 + 16 + akb];
    const int br0 = bufOff + (nw * 32 + lm) * 40;
    const int br1 = br0 + 16 * 40;
    fb0.q[0] = *(const U128*)&Bs[br0 + bkb];
    fb0.q[1] = *(const U128*)&Bs[br0 + bkb + 16];
    fb1.q[0] = *(const U128*)&Bs[br1 + bkb];
    fb1.q[1] = *(const U128*)&Bs[br1 + bkb + 16];
    a00 = __builtin_amdgcn_wmma_f32_16x16x32_bf16(false, fa0.v, false, fb0.v, (short)0, a00, false, false);
    a01 = __builtin_amdgcn_wmma_f32_16x16x32_bf16(false, fa0.v, false, fb1.v, (short)0, a01, false, false);
    a10 = __builtin_amdgcn_wmma_f32_16x16x32_bf16(false, fa1.v, false, fb0.v, (short)0, a10, false, false);
    a11 = __builtin_amdgcn_wmma_f32_16x16x32_bf16(false, fa1.v, false, fb1.v, (short)0, a11, false, false);
  }

  // ---- epilogue: bias / exact GELU / residual ----
  float* Cb = p.C + p.cOff + (long long)zb * p.cSB + (long long)zh * p.cSH;
  const float* Rb = (p.flags & 4)
      ? (p.R + p.rOff + (long long)zb * p.rSB + (long long)zh * p.rSH) : nullptr;
  const int rowAdd = (lane & 16) ? 8 : 0;
  const int lm = lane & 15;
  const f32x8* accs[4] = { &a00, &a01, &a10, &a11 };
  #pragma unroll
  for (int mi = 0; mi < 2; ++mi) {
    const int row0 = bm0 + mw * 32 + mi * 16 + rowAdd;
    long long cOffRow = (long long)row0 * p.ldc;
    long long rOffRow = (long long)row0 * p.ldr;
    #pragma unroll
    for (int rr = 0; rr < 8; ++rr, cOffRow += p.ldc, rOffRow += p.ldr) {
      const int row = row0 + rr;
      if (row >= p.M) continue;
      #pragma unroll
      for (int ni = 0; ni < 2; ++ni) {
        const int col = bn0 + nw * 32 + ni * 16 + lm;
        if (col >= p.N) continue;
        float v = (*accs[mi * 2 + ni])[rr];
        if (p.flags & 1) v += p.bias[col];
        if (p.flags & 2) v = 0.5f * v * (1.0f + erff(v * 0.70710678118654752f));
        if (p.flags & 4) v += Rb[rOffRow + col];
        Cb[cOffRow + col] = v;
      }
    }
  }
}

// ---------------- small kernels ----------------
// weights: W[K][N] f32 -> Wt[N][Kp] bf16 (transposed, zero-padded to Kp)
__global__ void wtT_k(const float* __restrict__ W, unsigned short* __restrict__ Wt,
                      int K, int N, int Kp, long long total) {
  long long idx = (long long)blockIdx.x * 256 + threadIdx.x;
  if (idx >= total) return;
  int k = (int)(idx % Kp); long long r = idx / Kp; int n = (int)r;
  Wt[idx] = (k < K) ? f2bf(W[(long long)k * N + n]) : (unsigned short)0;
}

// im2col: kernel 3x30, stride (1,15), C=3, W=300, ow=19, patch_dim=270
__global__ void im2col_k(const float* __restrict__ src, float* __restrict__ dst,
                         int Hh, int oh, long long total) {
  long long idx = (long long)blockIdx.x * 256 + threadIdx.x;
  if (idx >= total) return;
  int c = (int)(idx % 270); long long r = idx / 270;
  int P = oh * 19;
  int pp = (int)(r % P); int b = (int)(r / P);
  int ci = c / 90, rem = c % 90, kh = rem / 30, kw = rem % 30;
  int ph = pp / 19, pw = pp % 19;
  dst[idx] = src[(((long long)b * 3 + ci) * Hh + (ph + kh)) * 300 + (pw * 15 + kw)];
}

__global__ void tokens_k(const float* __restrict__ emb, const float* __restrict__ cls,
                         const float* __restrict__ pos, float* __restrict__ X,
                         int T, long long total) {
  long long idx = (long long)blockIdx.x * 256 + threadIdx.x;
  if (idx >= total) return;
  int e = (int)(idx % 768); long long r = idx / 768;
  int n = (int)(r % T); int b = (int)(r / T);
  float v = (n == 0) ? cls[e] : emb[((long long)b * (T - 1) + (n - 1)) * 768 + e];
  X[idx] = v + pos[(long long)n * 768 + e];
}

__global__ void final_k(const float* __restrict__ xf, const float* __restrict__ xb,
                        const float* __restrict__ clsF, float* __restrict__ XF,
                        long long total) {
  long long idx = (long long)blockIdx.x * 256 + threadIdx.x;
  if (idx >= total) return;
  int e = (int)(idx % 768); long long r = idx / 768;
  int n = (int)(r % 1407); int b = (int)(r / 1407);
  float v;
  if (n == 0)          v = clsF[e];
  else if (n <= 1159)  v = xf[((long long)b * 1160 + n) * 768 + e];
  else                 v = xb[((long long)b * 248 + (n - 1159)) * 768 + e];
  XF[idx] = v;
}

__global__ __launch_bounds__(256) void ln_k(const float* __restrict__ X,
                                            const float* __restrict__ g,
                                            const float* __restrict__ bta,
                                            float* __restrict__ O) {
  __shared__ float red[256];
  long long row = blockIdx.x;
  const float* xr = X + row * 768;
  int tid = threadIdx.x;
  float s = 0.f;
  for (int i = tid; i < 768; i += 256) s += xr[i];
  float mean = block_sum(s, red) * (1.0f / 768.0f);
  float vs = 0.f;
  for (int i = tid; i < 768; i += 256) { float d = xr[i] - mean; vs += d * d; }
  float var = block_sum(vs, red) * (1.0f / 768.0f);
  float inv = rsqrtf(var + 1e-5f);
  float* orow = O + row * 768;
  for (int i = tid; i < 768; i += 256)
    orow[i] = (xr[i] - mean) * inv * g[i] + bta[i];
}

// softmax over last dim then * 1/sqrt(768)  (matches reference exactly)
__global__ __launch_bounds__(256) void softmax_k(float* __restrict__ S, int T) {
  __shared__ float red[256];
  long long row = blockIdx.x;
  float* r = S + row * (long long)T;
  int tid = threadIdx.x;
  float mx = -3.0e38f;
  for (int i = tid; i < T; i += 256) mx = fmaxf(mx, r[i]);
  float m = block_max(mx, red);
  float s = 0.f;
  for (int i = tid; i < T; i += 256) { float e = expf(r[i] - m); r[i] = e; s += e; }
  float sum = block_sum(s, red);
  float sc = 0.03608439182435161f / sum;  // 1/sqrt(768)
  for (int i = tid; i < T; i += 256) r[i] *= sc;
}

__global__ __launch_bounds__(256) void head_k(const float* __restrict__ X, int T, int tok,
                                              const float* __restrict__ g,
                                              const float* __restrict__ bta,
                                              const float* __restrict__ w,
                                              const float* __restrict__ bb,
                                              float* __restrict__ out) {
  __shared__ float red[256];
  int b = blockIdx.x;
  const float* xr = X + ((long long)b * T + tok) * 768;
  int tid = threadIdx.x;
  float s = 0.f;
  for (int i = tid; i < 768; i += 256) s += xr[i];
  float mean = block_sum(s, red) * (1.0f / 768.0f);
  float vs = 0.f;
  for (int i = tid; i < 768; i += 256) { float d = xr[i] - mean; vs += d * d; }
  float var = block_sum(vs, red) * (1.0f / 768.0f);
  float inv = rsqrtf(var + 1e-5f);
  float d0 = 0.f, d1 = 0.f;
  for (int i = tid; i < 768; i += 256) {
    float nv = (xr[i] - mean) * inv * g[i] + bta[i];
    d0 += nv * w[i * 2 + 0];
    d1 += nv * w[i * 2 + 1];
  }
  float s0 = block_sum(d0, red);
  float s1 = block_sum(d1, red);
  if (tid == 0) { out[b * 2 + 0] = s0 + bb[0]; out[b * 2 + 1] = s1 + bb[1]; }
}

// ---------------- host orchestration ----------------
static void gemmL(hipStream_t st,
  const float* A, long long aOff, long long aSB, long long aSH, int sAr, int sAc,
  const float* Bm, long long bOff, long long bSB, long long bSH, int sBr, int sBc,
  const unsigned short* Wt, int Kp,
  const float* bias,
  const float* R, long long rOff, long long rSB, long long rSH, int ldr,
  float* C, long long cOff, long long cSB, long long cSH, int ldc,
  int M, int N, int K, int Hh, int nz, int flags) {
  GemmP p;
  p.A = A; p.aOff = aOff; p.aSB = aSB; p.aSH = aSH; p.sAr = sAr; p.sAc = sAc;
  p.Bm = Bm; p.bOff = bOff; p.bSB = bSB; p.bSH = bSH; p.sBr = sBr; p.sBc = sBc;
  p.Wt = Wt; p.Kp = Kp;
  p.bias = bias; p.R = R; p.rOff = rOff; p.rSB = rSB; p.rSH = rSH; p.ldr = ldr;
  p.C = C; p.cOff = cOff; p.cSB = cSB; p.cSH = cSH; p.ldc = ldc;
  p.M = M; p.N = N; p.K = K; p.H = Hh; p.flags = flags;
  dim3 g((unsigned)((N + 127) / 128), (unsigned)((M + 63) / 64), (unsigned)nz);
  wmma_gemm_k<<<g, 256, 0, st>>>(p);
}

struct LW { unsigned short *qkvT, *projT, *ff1T, *ff2T; };

// pr: ln1_g,ln1_b,qkv_w,qkv_b,proj_w,proj_b,ln2_g,ln2_b,ff1_w,ff1_b,ff2_w,ff2_b
static void run_block(const float* const* pr, const LW& w, float* X, int T,
                      float* hb, float* qk, float* big, hipStream_t st) {
  const int BT = 8 * T;
  ln_k<<<dim3((unsigned)BT), 256, 0, st>>>(X, pr[0], pr[1], hb);
  // QKV (bf16^T weights via double-buffered TDM)
  gemmL(st, hb, 0, 0, 0, 768, 1, nullptr, 0, 0, 0, 0, 0, w.qkvT, 768,
        pr[3], nullptr, 0, 0, 0, 0,
        qk, 0, 0, 0, 2304, BT, 2304, 768, 1, 1, 1 | 8);
  float* sc = big;
  for (int b = 0; b < 8; ++b) {
    long long qb = (long long)b * T * 2304;
    // energy[h,q,k] = Q.K (d=256), strided f32 gathers from packed qkv
    gemmL(st, qk, qb + 0, 0, 768, 2304, 3,
          qk, qb + 1, 0, 768, 3, 2304, nullptr, 0,
          nullptr, nullptr, 0, 0, 0, 0,
          sc, 0, 0, (long long)T * T, T, T, T, 256, 3, 3, 0);
    softmax_k<<<dim3((unsigned)(3 * T)), 256, 0, st>>>(sc, T);
    // out[q, h*256+dd] = att @ V -> hb as [BT,768]
    gemmL(st, sc, 0, 0, (long long)T * T, T, 1,
          qk, qb + 2, 0, 768, 2304, 3, nullptr, 0,
          nullptr, nullptr, 0, 0, 0, 0,
          hb, (long long)b * T * 768, 0, 256, 768, T, 256, T, 3, 3, 0);
  }
  gemmL(st, hb, 0, 0, 0, 768, 1, nullptr, 0, 0, 0, 0, 0, w.projT, 768,
        pr[5], X, 0, 0, 0, 768,
        X, 0, 0, 0, 768, BT, 768, 768, 1, 1, 1 | 4 | 8);
  ln_k<<<dim3((unsigned)BT), 256, 0, st>>>(X, pr[6], pr[7], hb);
  gemmL(st, hb, 0, 0, 0, 768, 1, nullptr, 0, 0, 0, 0, 0, w.ff1T, 768,
        pr[9], nullptr, 0, 0, 0, 0,
        big, 0, 0, 0, 3072, BT, 3072, 768, 1, 1, 1 | 2 | 8);
  gemmL(st, big, 0, 0, 0, 3072, 1, nullptr, 0, 0, 0, 0, 0, w.ff2T, 3072,
        pr[11], X, 0, 0, 0, 768,
        X, 0, 0, 0, 768, BT, 768, 3072, 1, 1, 1 | 4 | 8);
}

extern "C" void kernel_launch(void* const* d_in, const int* in_sizes, int n_in,
                              void* d_out, int out_size, void* d_ws, size_t ws_size,
                              hipStream_t stream) {
  (void)n_in; (void)out_size; (void)ws_size;
  auto F = [&](int i) { return (const float*)d_in[i]; };
  float* out = (float*)d_out;
  float* ws = (float*)d_ws;

  // --- input index mapping: insertion-order or sorted-key flattening ---
  int iX, iY, iPW, iPB, iCF, iPF, iCB, iPBg, iCFin, enc6B, enc1B, hF, hB, hFin;
  int perm[12], hperm[4];
  if (in_sizes[0] == 8 * 3 * 63 * 300) {  // insertion order
    iX = 0; iY = 1; iPW = 2; iPB = 3; iCF = 4; iPF = 5; iCB = 6; iPBg = 7; iCFin = 8;
    enc6B = 9; enc1B = 81; hF = 93; hB = 97; hFin = 101;
    for (int j = 0; j < 12; ++j) perm[j] = j;
    hperm[0] = 0; hperm[1] = 1; hperm[2] = 2; hperm[3] = 3;
  } else {                                 // sorted keys (jax pytree style)
    iCB = 0; iCF = 1; iCFin = 2; enc1B = 3; enc6B = 15;
    hB = 87; hF = 91; hFin = 95; iPB = 99; iPW = 100; iPBg = 101; iPF = 102; iX = 103; iY = 104;
    int pm[12] = {5, 4, 11, 10, 9, 8, 7, 6, 1, 0, 3, 2};
    for (int j = 0; j < 12; ++j) perm[j] = pm[j];
    hperm[0] = 2; hperm[1] = 1; hperm[2] = 3; hperm[3] = 0;
  }

  // --- workspace layout (floats), ~446 MB total ---
  size_t o = 0;
  auto al = [&](size_t n) { size_t r = o; o += (n + 63) & ~(size_t)63; return r; };
  float* xf  = ws + al((size_t)8 * 1160 * 768);
  float* xb  = ws + al((size_t)8 * 248 * 768);
  float* xF  = ws + al((size_t)8 * 1407 * 768);
  float* hb  = ws + al((size_t)8 * 1407 * 768);
  float* qk  = ws + al((size_t)8 * 1407 * 2304);
  float* big = ws + al((size_t)8 * 1407 * 3072);  // shared: im2col/emb, scores, ff buffer
  // bf16^T weight region (ushorts)
  size_t wUsh = (size_t)768 * 272 + 7ull * ((size_t)2304 * 768 + (size_t)768 * 768 +
                (size_t)3072 * 768 + (size_t)768 * 3072) + 4096;
  unsigned short* wreg = (unsigned short*)(ws + al((wUsh + 1) / 2));
  size_t wo = 0;
  auto wal = [&](size_t n) { size_t r = wo; wo += (n + 31) & ~(size_t)31; return r; };
  unsigned short* patchT = wreg + wal((size_t)768 * 272);
  LW lw[7];
  for (int l = 0; l < 7; ++l) {
    lw[l].qkvT = wreg + wal((size_t)2304 * 768);
    lw[l].projT = wreg + wal((size_t)768 * 768);
    lw[l].ff1T = wreg + wal((size_t)3072 * 768);
    lw[l].ff2T = wreg + wal((size_t)768 * 3072);
  }
  // pre-encoder sub-buffers inside `big`
  size_t so = 0;
  auto sal = [&](size_t n) { size_t r = so; so += (n + 63) & ~(size_t)63; return r; };
  float* pf  = big + sal((size_t)8 * 1159 * 270);
  float* pbg = big + sal((size_t)8 * 247 * 270);
  float* ef  = big + sal((size_t)8 * 1159 * 768);
  float* ebg = big + sal((size_t)8 * 247 * 768);

  // 0) convert weights to transposed bf16 (Kp = K rounded to 16)
  auto cvtW = [&](const float* W, unsigned short* Wt, int K, int N) {
    int Kp = (K + 15) & ~15;
    long long tot = (long long)N * Kp;
    wtT_k<<<dim3((unsigned)((tot + 255) / 256)), 256, 0, stream>>>(W, Wt, K, N, Kp, tot);
  };
  cvtW(F(iPW), patchT, 270, 768);
  for (int l = 0; l < 7; ++l) {
    int base = (l < 6) ? (enc6B + 12 * l) : enc1B;
    cvtW(F(base + perm[2]),  lw[l].qkvT, 768, 2304);
    cvtW(F(base + perm[4]),  lw[l].projT, 768, 768);
    cvtW(F(base + perm[8]),  lw[l].ff1T, 768, 3072);
    cvtW(F(base + perm[10]), lw[l].ff2T, 3072, 768);
  }

  // 1) im2col (unfold)
  { long long tot = 8LL * 1159 * 270;
    im2col_k<<<dim3((unsigned)((tot + 255) / 256)), 256, 0, stream>>>(F(iX), pf, 63, 61, tot); }
  { long long tot = 8LL * 247 * 270;
    im2col_k<<<dim3((unsigned)((tot + 255) / 256)), 256, 0, stream>>>(F(iY), pbg, 15, 13, tot); }
  // 2) patch embedding GEMMs (TDM/bf16 weights)
  gemmL(stream, pf, 0, 0, 0, 270, 1, nullptr, 0, 0, 0, 0, 0, patchT, 272,
        F(iPB), nullptr, 0, 0, 0, 0, ef, 0, 0, 0, 768, 8 * 1159, 768, 270, 1, 1, 1 | 8);
  gemmL(stream, pbg, 0, 0, 0, 270, 1, nullptr, 0, 0, 0, 0, 0, patchT, 272,
        F(iPB), nullptr, 0, 0, 0, 0, ebg, 0, 0, 0, 768, 8 * 247, 768, 270, 1, 1, 1 | 8);
  // 3) prepend cls, add positional embeddings
  { long long tot = 8LL * 1160 * 768;
    tokens_k<<<dim3((unsigned)((tot + 255) / 256)), 256, 0, stream>>>(ef, F(iCF), F(iPF), xf, 1160, tot); }
  { long long tot = 8LL * 248 * 768;
    tokens_k<<<dim3((unsigned)((tot + 255) / 256)), 256, 0, stream>>>(ebg, F(iCB), F(iPBg), xb, 248, tot); }
  // 4) shared 6-layer encoder on face and bg streams
  for (int l = 0; l < 6; ++l) {
    const float* pr[12];
    for (int j = 0; j < 12; ++j) pr[j] = F(enc6B + 12 * l + perm[j]);
    run_block(pr, lw[l], xf, 1160, hb, qk, big, stream);
  }
  for (int l = 0; l < 6; ++l) {
    const float* pr[12];
    for (int j = 0; j < 12; ++j) pr[j] = F(enc6B + 12 * l + perm[j]);
    run_block(pr, lw[l], xb, 248, hb, qk, big, stream);
  }
  // 5) fuse streams and run the 1-layer encoder
  { long long tot = 8LL * 1407 * 768;
    final_k<<<dim3((unsigned)((tot + 255) / 256)), 256, 0, stream>>>(xf, xb, F(iCFin), xF, tot); }
  {
    const float* pr[12];
    for (int j = 0; j < 12; ++j) pr[j] = F(enc1B + perm[j]);
    run_block(pr, lw[6], xF, 1407, hb, qk, big, stream);
  }
  // 6) heads (output order: final, face, bg; face/bg use token 1)
  head_k<<<8, 256, 0, stream>>>(xF, 1407, 0, F(hFin + hperm[0]), F(hFin + hperm[1]),
                                F(hFin + hperm[2]), F(hFin + hperm[3]), out + 0);
  head_k<<<8, 256, 0, stream>>>(xf, 1160, 1, F(hF + hperm[0]), F(hF + hperm[1]),
                                F(hF + hperm[2]), F(hF + hperm[3]), out + 16);
  head_k<<<8, 256, 0, stream>>>(xb, 248, 1, F(hB + hperm[0]), F(hB + hperm[1]),
                                F(hB + hperm[2]), F(hB + hperm[3]), out + 32);
}
